// PhysicsLoss_3307124818510
// MI455X (gfx1250) — compile-verified, matrix operand order checked by reading, not run on hardware
//
#include <hip/hip_runtime.h>
#include <hip/hip_bf16.h>

typedef __attribute__((ext_vector_type(16))) __bf16 v16bf;
typedef __attribute__((ext_vector_type(8)))  __bf16 v8bf;
typedef __attribute__((ext_vector_type(8)))  float  v8f;
typedef __attribute__((ext_vector_type(4)))  unsigned int u32x4;
typedef __attribute__((ext_vector_type(8)))  int i32x8;
typedef __attribute__((ext_vector_type(4)))  int i32x4;

#define H     256
#define MT    16            // points per workgroup
#define RS    264           // LDS row stride in bf16 elems (16B aligned, bank-skewed)
#define NSTR  7             // streams: v, d0..d2, s0..s2

// ---- dynamic LDS layout (bytes) ----
#define WT_OFF     0
#define WT_BYTES   (H * RS * 2)            // 135168  (transposed weights, bf16)
#define BIAS_OFF   (WT_OFF + WT_BYTES)
#define BIAS_BYTES (H * 4)                 // 1024
#define ACTA_OFF   (BIAS_OFF + BIAS_BYTES)
#define ACT_BYTES  (NSTR * MT * RS * 2)    // 59136
#define ACTB_OFF   (ACTA_OFF + ACT_BYTES)
#define RED_OFF    (ACTB_OFF + ACT_BYTES)
#define RED_BYTES  (MT * 24 * 4)           // 1536
#define SMEM_TOTAL (RED_OFF + RED_BYTES)   // 256000 B < 320 KB

#define WT_WS_BYTES (3ull * H * H * 2ull)  // 393216 B of d_ws for bf16 W^T[1..3]

// Load a 16x32 bf16 WMMA fragment row for this lane from an LDS matrix with
// row stride RS. Layout per ISA 7.12.2 (16-bit A 16x32; B mirrored):
//   lanes 0-15 : row = lane,     K = {kb..kb+7, kb+16..kb+23}
//   lanes 16-31: row = lane-16,  K subset shifted by +8
__device__ __forceinline__ v16bf load_frag(const __bf16* base, int row, int kbase) {
    const __bf16* p = base + row * RS + kbase;
    v8bf lo = *(const v8bf*)(p);        // ds_load_b128
    v8bf hi = *(const v8bf*)(p + 16);   // ds_load_b128
    v16bf r;
#pragma unroll
    for (int i = 0; i < 8; ++i) { r[i] = lo[i]; r[i + 8] = hi[i]; }
    return r;
}

// ---- one-time prep: W^T (bf16) for hidden layers 1..3 into workspace ----
// ws layout: wsWT[L][j*H + k] = (bf16) W_L[k*H + j]
__global__ __launch_bounds__(256, 1)
void prep_wt_kernel(const float* __restrict__ W1, const float* __restrict__ W2,
                    const float* __restrict__ W3, unsigned short* __restrict__ ws)
{
    const int b = blockIdx.x;     // 0..767
    const int L = b >> 8;
    const int k = b & 255;        // source row of W (input neuron)
    const int j = threadIdx.x;    // output neuron
    const float* W = (L == 0) ? W1 : (L == 1) ? W2 : W3;
    __bf16* wsWT = (__bf16*)ws;
    wsWT[(size_t)L * (H * H) + (size_t)j * H + k] = (__bf16)W[k * H + j];
}

// Issue one TDM descriptor: copy H x H bf16 tile (row stride H elems in
// memory) into LDS at `ldsOff`, padding 4 DWORDs after every 128 DWORDs
// (one 256-elem row) -> effective LDS row stride of RS=264 bf16.
__device__ __forceinline__ void tdm_load_wt(const __bf16* src, unsigned ldsOff) {
    const unsigned long long ga = (unsigned long long)(uintptr_t)src;
    u32x4 g0;
    g0[0] = 1u;                                        // count=1, user mode
    g0[1] = ldsOff;                                    // lds_addr
    g0[2] = (unsigned)(ga & 0xFFFFFFFFu);              // global_addr[31:0]
    g0[3] = (unsigned)((ga >> 32) & 0x01FFFFFFu)       // global_addr[56:32]
          | 0x80000000u;                               // type=2 ("image")
    i32x8 g1;
    g1[0] = (int)0x07910000;     // data_size=1(2B) | pad_en | interval=6 | amount=3
    g1[1] = (int)(256u << 16);   // tensor_dim0 = 256      (bits 79:48)
    g1[2] = (int)(256u << 16);   // tensor_dim1 = 256      (bits 111:80)
    g1[3] = (int)(256u << 16);   // tile_dim0   = 256      (bits 127:112)
    g1[4] = (int)256;            // tile_dim1   = 256      (bits 143:128)
    g1[5] = (int)256;            // tensor_dim0_stride=256 (bits 207:160)
    g1[6] = 0;
    g1[7] = 0;
    i32x4 z4 = {};
#if defined(__clang_major__) && (__clang_major__ >= 23)
    i32x8 z8 = {};
    __builtin_amdgcn_tensor_load_to_lds(g0, g1, z4, z4, z8, 0);
#else
    __builtin_amdgcn_tensor_load_to_lds(g0, g1, z4, z4, 0);
#endif
    __builtin_amdgcn_s_wait_tensorcnt(0);
}

__global__ __launch_bounds__(256, 1)
void pinn_phyloss_kernel(const float* __restrict__ x_norm,
                         const float* __restrict__ nu,
                         const float* __restrict__ x_std,
                         const float* __restrict__ y_std,
                         const float* __restrict__ W0, const float* __restrict__ b0,
                         const float* __restrict__ W1, const float* __restrict__ b1,
                         const float* __restrict__ W2, const float* __restrict__ b2,
                         const float* __restrict__ W3, const float* __restrict__ b3,
                         const float* __restrict__ W4, const float* __restrict__ b4,
                         const unsigned short* __restrict__ wsWT_raw, int useTdm,
                         float* __restrict__ out)
{
    (void)b4; // output bias does not affect derivatives
    extern __shared__ char smem[];
    __bf16* wt   = (__bf16*)(smem + WT_OFF);
    float*  bias = (float*) (smem + BIAS_OFF);
    __bf16* actA = (__bf16*)(smem + ACTA_OFF);
    __bf16* actB = (__bf16*)(smem + ACTB_OFF);
    float*  red  = (float*) (smem + RED_OFF);

    const int tid  = threadIdx.x;
    const int lane = tid & 31;
    const int wave = tid >> 5;
    const int p0g  = blockIdx.x * MT;

    // ---------------- layer 0: R^3 -> H, seed 7 jet streams ----------------
    {
        const int j = tid; // one output neuron per thread, all 16 points
        const float w0 = W0[0 * H + j], w1 = W0[1 * H + j], w2 = W0[2 * H + j];
        const float bb = b0[j];
#pragma unroll 1
        for (int m = 0; m < MT; ++m) {
            const float x0 = x_norm[(p0g + m) * 3 + 0];
            const float x1 = x_norm[(p0g + m) * 3 + 1];
            const float x2 = x_norm[(p0g + m) * 3 + 2];
            const float pre = bb + x0 * w0 + x1 * w1 + x2 * w2;
            const float t   = tanhf(pre);
            const float dt  = 1.0f - t * t;
            const float ddt = -2.0f * t * dt;
            actA[(0 * MT + m) * RS + j] = (__bf16)t;
            actA[(1 * MT + m) * RS + j] = (__bf16)(dt * w0);
            actA[(2 * MT + m) * RS + j] = (__bf16)(dt * w1);
            actA[(3 * MT + m) * RS + j] = (__bf16)(dt * w2);
            actA[(4 * MT + m) * RS + j] = (__bf16)(ddt * w0 * w0);
            actA[(5 * MT + m) * RS + j] = (__bf16)(ddt * w1 * w1);
            actA[(6 * MT + m) * RS + j] = (__bf16)(ddt * w2 * w2);
        }
    }

    const float* Ws[3] = {W1, W2, W3};
    const float* bs[3] = {b1, b2, b3};
    __bf16* actIn  = actA;
    __bf16* actOut = actB;

    // ---------------- hidden layers 1..3 via WMMA ----------------
#pragma unroll 1
    for (int L = 0; L < 3; ++L) {
        __syncthreads(); // previous layer's reads of wt/acts complete

        if (useTdm) {
            // Tensor Data Mover stages pre-converted bf16 W^T (from d_ws)
            // into LDS with hardware padding to row stride RS. One wave
            // issues the DMA; TENSORcnt gates the barrier.
            if (tid < 32) {
                const __bf16* src =
                    (const __bf16*)wsWT_raw + (size_t)L * (H * H);
                tdm_load_wt(src, (unsigned)(uintptr_t)wt);
            }
            bias[tid] = bs[L][tid];
        } else {
            // fallback: in-kernel convert+transpose staging
            const float* W = Ws[L];
#pragma unroll 1
            for (int k = 0; k < H; ++k)
                wt[tid * RS + k] = (__bf16)W[k * H + tid];
            bias[tid] = bs[L][tid];
        }
        __syncthreads();

        const int mhi   = (lane < 16) ? 0 : 8;
        const int ksoff = (lane < 16) ? 0 : 8;
        const v8f czero = {};

#pragma unroll 1
        for (int ti = 0; ti < 2; ++ti) {
            const int nt = wave + ti * 8;           // N-tile 0..15
            const int j  = nt * 16 + (lane & 15);   // this lane's output neuron

            v8f acc[NSTR];
#pragma unroll
            for (int s = 0; s < NSTR; ++s) acc[s] = czero;

#pragma unroll 1
            for (int ks = 0; ks < 8; ++ks) {
                const int kbase = ks * 32 + ksoff;
                const v16bf bfrag = load_frag(wt, j, kbase); // B: W^T row j
#pragma unroll
                for (int s = 0; s < NSTR; ++s) {
                    const v16bf afrag =
                        load_frag(actIn, s * MT + (lane & 15), kbase);
                    acc[s] = __builtin_amdgcn_wmma_f32_16x16x32_bf16(
                        false, afrag, false, bfrag, (short)0, acc[s],
                        false, false);
                }
            }

            // bias + tanh chain rule directly on fragments, write back bf16
            const float bj = bias[j];
#pragma unroll
            for (int r = 0; r < 8; ++r) {
                const int m   = r + mhi;
                const float t   = tanhf(acc[0][r] + bj);
                const float dt  = 1.0f - t * t;
                const float ddt = -2.0f * t * dt;
                actOut[(0 * MT + m) * RS + j] = (__bf16)t;
#pragma unroll
                for (int i = 0; i < 3; ++i) {
                    const float dl = acc[1 + i][r];
                    const float sl = acc[4 + i][r];
                    actOut[((1 + i) * MT + m) * RS + j] = (__bf16)(dt * dl);
                    actOut[((4 + i) * MT + m) * RS + j] =
                        (__bf16)(ddt * dl * dl + dt * sl);
                }
            }
        }
        // ping-pong
        __bf16* tsw = actIn; actIn = actOut; actOut = tsw;
    }

    __syncthreads();

    // ---------------- output layer (H -> 4) + residual assembly ----------------
    {
        const int m   = tid >> 4;
        const int sub = tid & 15;
        if (sub < 12) {
            const int y = sub / 3, i = sub % 3; // output comp y, input dir i
            const __bf16* drow = actIn + ((1 + i) * MT + m) * RS;
            const __bf16* srow = actIn + ((4 + i) * MT + m) * RS;
            float J = 0.0f, Hd = 0.0f;
#pragma unroll 1
            for (int kk = 0; kk < H; kk += 8) {
                const v8bf dv = *(const v8bf*)(drow + kk);
                const v8bf sv = *(const v8bf*)(srow + kk);
#pragma unroll
                for (int q = 0; q < 8; ++q) {
                    const float w = W4[(kk + q) * 4 + y];
                    J  += (float)dv[q] * w;
                    Hd += (float)sv[q] * w;
                }
            }
            const float ys = y_std[y], xs = x_std[i];
            red[m * 24 + sub]      = J  * ys / xs;         // scaled Jacobian
            red[m * 24 + 12 + sub] = Hd * ys / (xs * xs);  // scaled Hess diag
        }
    }
    __syncthreads();

    if (tid < MT) {
        const float* Jm = red + tid * 24;  // Jm[y*3+i]
        const float* Hm = Jm + 12;
        const float nus  = nu[0];
        const float cont = Jm[0 * 3 + 0] + Jm[1 * 3 + 1] + Jm[2 * 3 + 2];
        const float lap0 = Hm[0] + Hm[1] + Hm[2];
        const float lap1 = Hm[3] + Hm[4] + Hm[5];
        const float lap2 = Hm[6] + Hm[7] + Hm[8];
        const float mx = Jm[3 * 3 + 0] - nus * lap0;
        const float my = Jm[3 * 3 + 1] - nus * lap1;
        const float mz = Jm[3 * 3 + 2] - nus * lap2;
        out[p0g + tid] = cont * cont + mx * mx + my * my + mz * mz;
    }
}

extern "C" void kernel_launch(void* const* d_in, const int* in_sizes, int n_in,
                              void* d_out, int out_size, void* d_ws, size_t ws_size,
                              hipStream_t stream) {
    (void)n_in; (void)out_size;
    const float* x_norm = (const float*)d_in[0];
    const float* nu     = (const float*)d_in[1];
    const float* x_std  = (const float*)d_in[2];
    const float* y_std  = (const float*)d_in[3];
    const float* W0 = (const float*)d_in[4];  const float* b0 = (const float*)d_in[5];
    const float* W1 = (const float*)d_in[6];  const float* b1 = (const float*)d_in[7];
    const float* W2 = (const float*)d_in[8];  const float* b2 = (const float*)d_in[9];
    const float* W3 = (const float*)d_in[10]; const float* b3 = (const float*)d_in[11];
    const float* W4 = (const float*)d_in[12]; const float* b4 = (const float*)d_in[13];
    float* out = (float*)d_out;

    const int npts   = in_sizes[0] / 3;   // 32768
    const int blocks = npts / MT;         // 2048

    // Deterministic path choice: pure function of the harness-fixed ws_size.
    const int useTdm = (d_ws != nullptr && ws_size >= (size_t)WT_WS_BYTES) ? 1 : 0;
    if (useTdm) {
        prep_wt_kernel<<<3 * H, 256, 0, stream>>>(W1, W2, W3,
                                                  (unsigned short*)d_ws);
    }

    (void)hipFuncSetAttribute((const void*)pinn_phyloss_kernel,
                              hipFuncAttributeMaxDynamicSharedMemorySize,
                              SMEM_TOTAL);
    pinn_phyloss_kernel<<<blocks, 256, SMEM_TOTAL, stream>>>(
        x_norm, nu, x_std, y_std,
        W0, b0, W1, b1, W2, b2, W3, b3, W4, b4,
        (const unsigned short*)d_ws, useTdm, out);
}
